// CustomPeepholeLSTM_31997506356095
// MI455X (gfx1250) — matrix operand order repaired and checked by reading, not compiled, not run on hardware
//
#include <hip/hip_runtime.h>
#include <hip/hip_bf16.h>

typedef __attribute__((ext_vector_type(16))) __bf16 v16bf;
typedef __attribute__((ext_vector_type(8)))  float  v8f;

#define B_    128
#define S_    1024
#define HS_   256
#define G4_   1024          // 4*HS
#define NWG   32            // workgroups (each owns 8 hidden columns)
#define COLS  8             // hidden columns per WG
#define NT    256           // threads per block = 8 wave32

// ---------------------------------------------------------------------------
// init: zero both h double-buffers (h0 = 0) and the grid-barrier counter.
// Runs every launch so graph replays are deterministic.
// ---------------------------------------------------------------------------
__global__ void lstm_init_kernel(unsigned* __restrict__ hbuf_u,
                                 unsigned* __restrict__ cnt) {
  int idx = blockIdx.x * blockDim.x + threadIdx.x;
  // 2 buffers * 128 * 256 bf16 = 32768 dwords
  for (int i = idx; i < 2 * B_ * HS_ / 2; i += gridDim.x * blockDim.x)
    hbuf_u[i] = 0u;
  if (idx == 0) *cnt = 0u;
}

// ---------------------------------------------------------------------------
// Persistent fused LSTM. One WG = 8 hidden columns = 32 gate columns.
// gates[B,32] = [x_t | h] (128x512, bf16) @ [W;U] slice (512x32, bf16) + bias
// computed as 8 waves x (16-row tile x 2 16-col WMMA tiles) x 16 K-chunks.
// ---------------------------------------------------------------------------
__global__ void __launch_bounds__(NT)
lstm_persistent_kernel(const float* __restrict__ x,     // [B,S,I] f32
                       const float* __restrict__ W,     // [I,4HS] f32
                       const float* __restrict__ U,     // [HS,4HS] f32
                       const float* __restrict__ bias,  // [4HS] f32
                       float* __restrict__ out,         // [B,S,HS] + [B,HS] + [B,HS]
                       __bf16* __restrict__ hbuf,       // [2][B][HS] bf16 (d_ws)
                       unsigned* __restrict__ cnt)      // grid barrier counter
{
  // Weights in per-lane WMMA B-fragment order: slot (kc,tt,lane) -> 32 bytes
  __shared__ __align__(16) unsigned char ldsW[16 * 2 * 32 * 32];  // 32 KB
  __shared__ float ldsG[B_ * 32];                                  // 16 KB gate tiles
  __shared__ float ldsC[B_ * COLS];                                //  4 KB cell state

  const int tid  = threadIdx.x;
  const int wg   = blockIdx.x;          // 0..31
  const int lane = tid & 31;
  const int wave = tid >> 5;            // 0..7
  const int Mbase = wave * 16;          // rows (batch) owned by this wave

  // ---- one-time: swizzle W/U slice into B-fragment order in LDS ----------
  // B 32x16 bf16 layout: lane n (N=lane%16) holds K = kbase + (lane<16?0:8) + {0-7,16-23}
  for (int s = tid; s < 16 * 2 * 32; s += NT) {
    const int kc   = s >> 6;            // K-chunk 0..15
    const int tt   = (s >> 5) & 1;      // N-tile 0..1
    const int ln   = s & 31;
    const int koff = (ln < 16) ? 0 : 8;
    const int nl   = ln & 15;
    // tile0 packs [i(8) | f(8)], tile1 packs [g(8) | o(8)]
    const int col  = (tt * 2 + (nl >> 3)) * HS_ + wg * COLS + (nl & 7);
    __bf16 tmp[16];
#pragma unroll
    for (int q = 0; q < 16; ++q) {
      const int k = kc * 32 + koff + (q & 7) + (q >> 3) * 16;  // concat-K: 0..255=x, 256..511=h
      const float v = (k < HS_) ? W[(size_t)k * G4_ + col]
                                : U[(size_t)(k - HS_) * G4_ + col];
      tmp[q] = (__bf16)v;
    }
    uint4* dst = (uint4*)(ldsW + (size_t)s * 32);
    dst[0] = ((const uint4*)tmp)[0];
    dst[1] = ((const uint4*)tmp)[1];
  }
  for (int s = tid; s < B_ * COLS; s += NT) ldsC[s] = 0.0f;

  // per-lane bias (folded into accumulator init; constant per column)
  const int nl = lane & 15;
  const float bias_t0 = bias[(0 + (nl >> 3)) * HS_ + wg * COLS + (nl & 7)];
  const float bias_t1 = bias[(2 + (nl >> 3)) * HS_ + wg * COLS + (nl & 7)];

  // A-fragment addressing (bf16 16x32): lanes<16 -> K {0-7,16-23}, lanes>=16 -> {8-15,24-31}
  const int arow  = Mbase + (lane & 15);
  const int akoff = (lane < 16) ? 0 : 8;

  __syncthreads();

  for (int t = 0; t < S_; ++t) {
    const __bf16* __restrict__ hin  = hbuf + (size_t)(t & 1) * (B_ * HS_);
    __bf16* __restrict__       hout = hbuf + (size_t)((t + 1) & 1) * (B_ * HS_);

    v8f acc0, acc1;
#pragma unroll
    for (int r = 0; r < 8; ++r) { acc0[r] = bias_t0; acc1[r] = bias_t1; }

#pragma unroll
    for (int kc = 0; kc < 16; ++kc) {
      // ---- A fragment (shared by both N-tiles) ----
      v16bf a;
      if (kc < 8) {
        // from x_t (f32, L2-resident): two 32B groups, convert to bf16
        const float* ap = x + ((size_t)arow * S_ + t) * HS_ + kc * 32 + akoff;
        const v8f lo = *(const v8f*)(ap);
        const v8f hi = *(const v8f*)(ap + 16);
#pragma unroll
        for (int q = 0; q < 8; ++q) {
          a[q]     = (__bf16)lo[q];
          a[8 + q] = (__bf16)hi[q];
        }
      } else {
        // from previous h (already bf16): two 16B loads
        const __bf16* ap = hin + (size_t)arow * HS_ + (kc - 8) * 32 + akoff;
        union { uint4 u[2]; v16bf v; } cv;
        cv.u[0] = *(const uint4*)(ap);
        cv.u[1] = *(const uint4*)(ap + 16);
        a = cv.v;
      }
      // ---- B fragments from LDS (pre-swizzled) ----
      const uint4* b0p = (const uint4*)(ldsW + (((size_t)kc * 2 + 0) * 32 + lane) * 32);
      const uint4* b1p = (const uint4*)(ldsW + (((size_t)kc * 2 + 1) * 32 + lane) * 32);
      union { uint4 u[2]; v16bf v; } b0, b1;
      b0.u[0] = b0p[0]; b0.u[1] = b0p[1];
      b1.u[0] = b1p[0]; b1.u[1] = b1p[1];

      acc0 = __builtin_amdgcn_wmma_f32_16x16x32_bf16(false, a, false, b0.v,
                                                     (short)0, acc0, false, false);
      acc1 = __builtin_amdgcn_wmma_f32_16x16x32_bf16(false, a, false, b1.v,
                                                     (short)0, acc1, false, false);
    }

    // ---- scatter C tiles to LDS gate buffer [128 x 32] ----
    // C layout: VGPR r, lanes0-15: M=r,N=lane; lanes16-31: M=r+8,N=lane-16
    {
      const int crow = Mbase + ((lane < 16) ? 0 : 8);
      const int ccol = lane & 15;
#pragma unroll
      for (int r = 0; r < 8; ++r) {
        ldsG[(crow + r) * 32 + ccol]      = acc0[r];
        ldsG[(crow + r) * 32 + 16 + ccol] = acc1[r];
      }
    }
    __syncthreads();

    // ---- elementwise cell update: 1024 elements, 4 per thread ----
#pragma unroll
    for (int p = 0; p < 4; ++p) {
      const int e  = tid + p * NT;   // 0..1023
      const int b  = e >> 3;         // batch row
      const int jl = e & 7;          // local hidden col
      const float gi = ldsG[b * 32 + jl];
      const float gf = ldsG[b * 32 + 8 + jl];
      const float gg = ldsG[b * 32 + 16 + jl];
      const float go = ldsG[b * 32 + 24 + jl];
      const float it = 1.0f / (1.0f + __expf(-gi));
      const float ft = 1.0f / (1.0f + __expf(-gf));
      const float gt = tanhf(gg);
      const float ot = 1.0f / (1.0f + __expf(-go));
      const float cprev = ldsC[b * COLS + jl];
      const float cnew  = ft * cprev + it * gt;
      const float hv    = ot * tanhf(cnew);
      ldsC[b * COLS + jl] = cnew;
      const int col = wg * COLS + jl;
      out[((size_t)b * S_ + t) * HS_ + col] = hv;          // hidden_seq
      hout[(size_t)b * HS_ + col] = (__bf16)hv;            // h for next step
      if (t == S_ - 1) {
        out[(size_t)B_ * S_ * HS_ + (size_t)b * HS_ + col]            = hv;    // h_t
        out[(size_t)B_ * S_ * HS_ + B_ * HS_ + (size_t)b * HS_ + col] = cnew;  // c_t
      }
    }

    // ---- device-scope grid barrier (release -> arrive -> acquire-spin) ----
    __threadfence();            // release our hout stores to DEV scope
    __syncthreads();            // everyone in WG has fenced
    if (tid == 0) {
      atomicAdd(cnt, 1u);
      const unsigned target = (unsigned)(t + 1) * (unsigned)NWG;
      while (__hip_atomic_load(cnt, __ATOMIC_ACQUIRE,
                               __HIP_MEMORY_SCOPE_AGENT) < target) {
        __builtin_amdgcn_s_sleep(1);
      }
    }
    __syncthreads();
    __threadfence();            // acquire: invalidate stale h lines in WGP$
  }
}

// ---------------------------------------------------------------------------
extern "C" void kernel_launch(void* const* d_in, const int* in_sizes, int n_in,
                              void* d_out, int out_size, void* d_ws, size_t ws_size,
                              hipStream_t stream) {
  (void)in_sizes; (void)n_in; (void)out_size; (void)ws_size;
  const float* x    = (const float*)d_in[0];   // [B,S,I]
  const float* W    = (const float*)d_in[1];   // [I,4HS]
  const float* U    = (const float*)d_in[2];   // [HS,4HS]
  const float* bias = (const float*)d_in[3];   // [4HS]
  float* out = (float*)d_out;

  __bf16*   hbuf = (__bf16*)d_ws;                                   // 2*B*HS bf16
  unsigned* cnt  = (unsigned*)((char*)d_ws + (size_t)2 * B_ * HS_ * sizeof(__bf16));

  lstm_init_kernel<<<32, NT, 0, stream>>>((unsigned*)d_ws, cnt);
  lstm_persistent_kernel<<<NWG, NT, 0, stream>>>(x, W, U, bias, out, hbuf, cnt);
}
